// SelfAttention_13348758356330
// MI455X (gfx1250) — matrix operand order repaired and checked by reading, not compiled, not run on hardware
//
#include <hip/hip_runtime.h>

// ---------- types ----------
typedef __attribute__((ext_vector_type(16))) __bf16 v16bf;
typedef __attribute__((ext_vector_type(8)))  float  v8f;

#define B_  4
#define S_  4096
#define D_  256
#define SCALE 0.0625f   // 1/sqrt(256)

union FragBits {
    struct { uint4 lo, hi; } u;
    v16bf v;
};

// 16 consecutive bf16 (32B) -> one fragment (B-frag pattern: K = 16*hi + i)
__device__ __forceinline__ v16bf load_bfrag(const __bf16* __restrict__ p) {
    FragBits f;
    f.u.lo = *(const uint4*)(p);
    f.u.hi = *(const uint4*)(p + 8);
    return f.v;
}

// two groups of 8 consecutive bf16 (A-frag pattern, p1 = p0 + 8: K = 16*hi + i)
__device__ __forceinline__ v16bf load_afrag(const __bf16* __restrict__ p0,
                                            const __bf16* __restrict__ p1) {
    FragBits f;
    f.u.lo = *(const uint4*)(p0);
    f.u.hi = *(const uint4*)(p1);
    return f.v;
}

// 16 consecutive fp32 -> bf16 fragment
__device__ __forceinline__ v16bf cvt_frag16(const float* __restrict__ p) {
    v16bf v;
#pragma unroll
    for (int i = 0; i < 16; ++i) v[i] = (__bf16)p[i];
    return v;
}

// two groups of 8 consecutive fp32 -> bf16 fragment
__device__ __forceinline__ v16bf cvt_frag8x2(const float* __restrict__ p0,
                                             const float* __restrict__ p1) {
    v16bf v;
#pragma unroll
    for (int i = 0; i < 8; ++i) { v[i] = (__bf16)p0[i]; v[i + 8] = (__bf16)p1[i]; }
    return v;
}

__device__ __forceinline__ v8f wmma_bf16(v16bf a, v16bf b, v8f c) {
    return __builtin_amdgcn_wmma_f32_16x16x32_bf16(
        /*neg_a=*/false, a, /*neg_b=*/false, b,
        /*c_mod=*/(short)0, c, /*reuse_a=*/false, /*reuse_b=*/false);
}

// pack 8 accumulator floats as bf16 and store 16B
__device__ __forceinline__ void store_pk8(__bf16* __restrict__ p, v8f a) {
    union { uint4 u; __bf16 h[8]; } o;
#pragma unroll
    for (int i = 0; i < 8; ++i) o.h[i] = (__bf16)a[i];
    *(uint4*)p = o.u;
}

// ============================================================================
// Kernel 0: projections. Q[s,o]=x·Wqᵀ, K likewise (row-major bf16),
//           V stored transposed: Vt[v][s].
// One wave (= one 32-thread block) per (matrix, 16-row s-tile); 3072 blocks.
// Q/K use form D=Wrow × xᵀ (D: M=o,N=s) so stores pack 8 contiguous o.
// V  uses form D=x × Wvᵀ   (D: M=s,N=v) so Vt stores pack 8 contiguous s.
// ============================================================================
__global__ __launch_bounds__(32) void proj_kernel(
                            const float* __restrict__ x,
                            const float* __restrict__ Wq,
                            const float* __restrict__ Wk,
                            const float* __restrict__ Wv,
                            __bf16* __restrict__ Qb,
                            __bf16* __restrict__ Kb,
                            __bf16* __restrict__ Vt) {
    const int lane = threadIdx.x & 31;
    const int task = blockIdx.x;                      // 0..3071
    const int ST   = (B_ * S_) / 16;                  // 1024 s-tiles
    const int midx = task / ST;                       // 0=Q 1=K 2=V
    const int st   = task % ST;
    const int b    = st / (S_ / 16);
    const int s0   = (st % (S_ / 16)) * 16;
    const int ln   = lane & 15;
    const int hi   = lane >> 4;

    const float* xb = x + ((size_t)b * S_ + s0) * D_;

    if (midx < 2) {
        const float* W  = (midx == 0) ? Wq : Wk;
        __bf16* Out     = ((midx == 0) ? Qb : Kb) + ((size_t)b * S_ + s0) * D_;
        // cache B-frags of xᵀ: lane -> s = s0+ln, frag f covers d in [32f,32f+32)
        v16bf bx[8];
#pragma unroll
        for (int f = 0; f < 8; ++f)
            bx[f] = cvt_frag16(xb + (size_t)ln * D_ + 32 * f + 16 * hi);
#pragma unroll
        for (int n = 0; n < 16; ++n) {
            v8f acc = {};
#pragma unroll
            for (int f = 0; f < 8; ++f) {
                const float* wp = W + (size_t)(16 * n + ln) * D_ + 32 * f + 16 * hi;
                acc = wmma_bf16(cvt_frag8x2(wp, wp + 8), bx[f], acc);
            }
            // D: M=o=16n+r+8hi, N=s=s0+ln -> row s, cols 16n+8hi..+7
            store_pk8(Out + (size_t)ln * D_ + 16 * n + 8 * hi, acc);
        }
    } else {
        __bf16* OutT = Vt + (size_t)b * D_ * S_;      // [v][s]
        // cache A-frags of x: lane -> s = s0+ln
        v16bf ax[8];
#pragma unroll
        for (int f = 0; f < 8; ++f) {
            const float* xp = xb + (size_t)ln * D_ + 32 * f + 16 * hi;
            ax[f] = cvt_frag8x2(xp, xp + 8);
        }
#pragma unroll
        for (int n = 0; n < 16; ++n) {
            v8f acc = {};
#pragma unroll
            for (int f = 0; f < 8; ++f)
                acc = wmma_bf16(ax[f],
                                cvt_frag16(Wv + (size_t)(16 * n + ln) * D_ + 32 * f + 16 * hi),
                                acc);
            // D: M=s=s0+r+8hi, N=v=16n+ln -> Vt row v, cols s0+8hi..+7
            store_pk8(OutT + (size_t)(16 * n + ln) * S_ + s0 + 8 * hi, acc);
        }
    }
}

// ============================================================================
// Kernel 1: per-key softmax stats over the query axis.
// One wave (block) owns 16 key columns; streams all 256 q-tiles. In the score
// D-tile (A=Q M=q, B=Kᵀ N=k) each lane holds a single key column (8 q/tile),
// so online max/sumexp is per-lane; lanes L and L^16 merge at the end.
// ============================================================================
__global__ __launch_bounds__(32) void stats_kernel(
                             const __bf16* __restrict__ Qb,
                             const __bf16* __restrict__ Kb,
                             float* __restrict__ mbuf,
                             float* __restrict__ rbuf) {
    const int lane = threadIdx.x & 31;
    const int task = blockIdx.x;                      // 0..1023
    const int b    = task / (S_ / 16);
    const int k0   = (task % (S_ / 16)) * 16;
    const int ln   = lane & 15;
    const int hi   = lane >> 4;

    const __bf16* Kp = Kb + ((size_t)b * S_ + k0) * D_;
    const __bf16* Qp = Qb + (size_t)b * S_ * D_;

    v16bf bk[8];                                      // cached Kᵀ B-frags
#pragma unroll
    for (int f = 0; f < 8; ++f)
        bk[f] = load_bfrag(Kp + (size_t)ln * D_ + 32 * f + 16 * hi);

    float m_run = -1e30f, s_run = 0.f;
    for (int qt = 0; qt < S_ / 16; ++qt) {
        v8f acc = {};
#pragma unroll
        for (int f = 0; f < 8; ++f) {
            const __bf16* qp = Qp + (size_t)(qt * 16 + ln) * D_ + 32 * f + 16 * hi;
            acc = wmma_bf16(load_afrag(qp, qp + 8), bk[f], acc);
        }
        float tm = -1e30f;
#pragma unroll
        for (int r = 0; r < 8; ++r) tm = fmaxf(tm, acc[r] * SCALE);
        float m_new = fmaxf(m_run, tm);
        float ssum = 0.f;
#pragma unroll
        for (int r = 0; r < 8; ++r) ssum += __expf(acc[r] * SCALE - m_new);
        s_run = s_run * __expf(m_run - m_new) + ssum;
        m_run = m_new;
    }
    float m_o = __shfl_xor(m_run, 16, 32);
    float s_o = __shfl_xor(s_run, 16, 32);
    float M   = fmaxf(m_run, m_o);
    float Sv  = s_run * __expf(m_run - M) + s_o * __expf(m_o - M);
    if (hi == 0) {
        mbuf[(size_t)b * S_ + k0 + ln] = M;
        rbuf[(size_t)b * S_ + k0 + ln] = 1.0f / Sv;
    }
}

// ============================================================================
// Kernel 2: out[q,:] = Σ_k exp(s[q,k]-m[k])·(1/Z[k])·V[k,:]
// One wave (block) owns a 16-query strip and ALL 256 output columns
// (16 v8f accums). Per 32-key step: 16 score WMMAs (Q frags cached in VGPRs),
// exp/scale, 16x32 P tile through LDS (D-layout -> A-layout), 16 P·Vᵀ WMMAs.
// ============================================================================
__global__ __launch_bounds__(32) void attn_out_kernel(
                                const __bf16* __restrict__ Qb,
                                const __bf16* __restrict__ Kb,
                                const __bf16* __restrict__ Vt,
                                const float* __restrict__ mbuf,
                                const float* __restrict__ rbuf,
                                float* __restrict__ out) {
    __shared__ __bf16 ldsP[16][32];                   // 1 KB, wave-private tile
    const int lane = threadIdx.x & 31;
    const int task = blockIdx.x;                      // 0..1023
    const int b    = task / (S_ / 16);
    const int q0   = (task % (S_ / 16)) * 16;
    const int ln   = lane & 15;
    const int hi   = lane >> 4;

    const __bf16* Qp = Qb + ((size_t)b * S_ + q0) * D_;
    const __bf16* Kp = Kb + (size_t)b * S_ * D_;
    const __bf16* Vp = Vt + (size_t)b * D_ * S_;
    const float*  mp = mbuf + (size_t)b * S_;
    const float*  rp = rbuf + (size_t)b * S_;

    v16bf qf[8];                                      // cached Q A-frags
#pragma unroll
    for (int f = 0; f < 8; ++f) {
        const __bf16* qp = Qp + (size_t)ln * D_ + 32 * f + 16 * hi;
        qf[f] = load_afrag(qp, qp + 8);
    }

    v8f acc[16];
#pragma unroll
    for (int n = 0; n < 16; ++n) acc[n] = (v8f){};

    for (int kk = 0; kk < S_ / 32; ++kk) {
        const int k0 = kk * 32;
        v8f sc0 = {}, sc1 = {};
#pragma unroll
        for (int f = 0; f < 8; ++f) {
            const __bf16* kp0 = Kp + (size_t)(k0 + ln) * D_ + 32 * f + 16 * hi;
            const __bf16* kp1 = Kp + (size_t)(k0 + 16 + ln) * D_ + 32 * f + 16 * hi;
            sc0 = wmma_bf16(qf[f], load_bfrag(kp0), sc0);
            sc1 = wmma_bf16(qf[f], load_bfrag(kp1), sc1);
        }
        const float m0 = mp[k0 + ln],      r0 = rp[k0 + ln];
        const float m1 = mp[k0 + 16 + ln], r1 = rp[k0 + 16 + ln];
        // D layout: lane column k=ln (sc0) / k=16+ln (sc1), rows q=r+8hi
#pragma unroll
        for (int r = 0; r < 8; ++r) {
            ldsP[8 * hi + r][ln]      = (__bf16)(__expf(sc0[r] * SCALE - m0) * r0);
            ldsP[8 * hi + r][ln + 16] = (__bf16)(__expf(sc1[r] * SCALE - m1) * r1);
        }
        asm volatile("s_wait_dscnt 0" ::: "memory");  // same-wave cross-lane via LDS
        const __bf16* prow = &ldsP[ln][16 * hi];
        v16bf pf = load_afrag(prow, prow + 8);        // P as A-frag (16q x 32k)
#pragma unroll
        for (int n = 0; n < 16; ++n) {
            const __bf16* vp = Vp + (size_t)(16 * n + ln) * S_ + k0 + 16 * hi;
            acc[n] = wmma_bf16(pf, load_bfrag(vp), acc[n]);
        }
    }

    float* op = out + ((size_t)b * S_ + q0) * D_;
#pragma unroll
    for (int n = 0; n < 16; ++n)
#pragma unroll
        for (int r = 0; r < 8; ++r)
            op[(size_t)(8 * hi + r) * D_ + 16 * n + ln] = acc[n][r];
}

// ============================================================================
extern "C" void kernel_launch(void* const* d_in, const int* in_sizes, int n_in,
                              void* d_out, int out_size, void* d_ws, size_t ws_size,
                              hipStream_t stream) {
    const float* x  = (const float*)d_in[0];
    const float* Wq = (const float*)d_in[1];
    const float* Wk = (const float*)d_in[2];
    const float* Wv = (const float*)d_in[3];
    float* out = (float*)d_out;

    // workspace layout (bf16 Q,K,Vt + fp32 stats) ~24.1 MB
    const size_t nQKV = (size_t)B_ * S_ * D_;         // 4,194,304 elements
    __bf16* Qb = (__bf16*)d_ws;
    __bf16* Kb = Qb + nQKV;
    __bf16* Vt = Kb + nQKV;
    float*  mb = (float*)(Vt + nQKV);
    float*  rb = mb + (size_t)B_ * S_;

    // one wave per block so the scheduler can spread work across every WGP
    proj_kernel    <<<dim3(3072), dim3(32), 0, stream>>>(x, Wq, Wk, Wv, Qb, Kb, Vt);
    stats_kernel   <<<dim3(1024), dim3(32), 0, stream>>>(Qb, Kb, mb, rb);
    attn_out_kernel<<<dim3(1024), dim3(32), 0, stream>>>(Qb, Kb, Vt, mb, rb, out);
}
